// TemporalFlowCell_91311004713373
// MI455X (gfx1250) — matrix operand order repaired and checked by reading, not compiled, not run on hardware
//
#include <hip/hip_runtime.h>
#include <hip/hip_bf16.h>
#include <math.h>

// ---------------------------------------------------------------------------
// TemporalFlowCell forward for MI455X (gfx1250, wave32, WMMA bf16,
// async global->LDS staging, double-buffered tiles)
// B=8, T=4096, D=1024, K=256, CHUNK=128
// ---------------------------------------------------------------------------

#define BB 8
#define TT 4096
#define DD 1024
#define KK 256
#define CHUNKN 128
#define NCHUNK (TT / CHUNKN)
#define MROWS (BB * TT)          // 32768 flattened (b,t) rows

typedef __attribute__((ext_vector_type(16))) __bf16 v16bf;
typedef __attribute__((ext_vector_type(8)))  float  v8f;

union FragBF { unsigned int u[8]; v16bf v; };
union AccF   { v8f v; float f[8]; };

static __device__ __forceinline__ unsigned short f32_to_bf16(float f) {
  union { float f; unsigned int u; } x; x.f = f;
  unsigned int r = x.u + 0x7FFFu + ((x.u >> 16) & 1u);   // round-to-nearest-even
  return (unsigned short)(r >> 16);
}

// ---------------------------------------------------------------------------
// ||M||_F : single-block reduction (K*K = 65536 elements, 256 per thread)
// ---------------------------------------------------------------------------
__global__ __launch_bounds__(256) void tfc_norm_kernel(const float* __restrict__ M,
                                                       float* __restrict__ norm_out) {
  __shared__ float red[256];
  float s = 0.f;
  for (int i = threadIdx.x; i < KK * KK; i += 256) { float v = M[i]; s += v * v; }
  red[threadIdx.x] = s;
  __syncthreads();
  for (int w = 128; w > 0; w >>= 1) {
    if (threadIdx.x < w) red[threadIdx.x] += red[threadIdx.x + w];
    __syncthreads();
  }
  if (threadIdx.x == 0) norm_out[0] = sqrtf(red[0]);
}

// R = I + 0.01 * M / (||M|| + 1e-8), produce f32 (carry GEMV) and bf16 (WMMA)
__global__ __launch_bounds__(256) void tfc_buildR_kernel(const float* __restrict__ M,
                                                         const float* __restrict__ norm,
                                                         float* __restrict__ Rf,
                                                         unsigned short* __restrict__ Rb) {
  int idx = blockIdx.x * 256 + threadIdx.x;           // grid = 256 blocks
  int i = idx >> 8, j = idx & 255;
  float scale = 0.01f / (norm[0] + 1e-8f);
  float v = ((i == j) ? 1.0f : 0.0f) + M[idx] * scale;
  Rf[idx] = v;
  Rb[idx] = f32_to_bf16(v);
}

// ---------------------------------------------------------------------------
// f32 -> bf16 grid-stride convert (for x)
// ---------------------------------------------------------------------------
__global__ __launch_bounds__(256) void tfc_cvt_kernel(const float* __restrict__ in,
                                                      unsigned short* __restrict__ out,
                                                      long long n) {
  long long i = (long long)blockIdx.x * 256 + threadIdx.x;
  long long stride = (long long)gridDim.x * 256;
  for (; i < n; i += stride) out[i] = f32_to_bf16(in[i]);
}

// transpose + convert: out[c*rows + r] = in[r*cols + c]   (in: rows x cols)
__global__ __launch_bounds__(256) void tfc_cvtT_kernel(const float* __restrict__ in,
                                                       unsigned short* __restrict__ out,
                                                       int rows, int cols) {
  int c = blockIdx.x * 256 + threadIdx.x;
  if (c >= cols) return;
  for (int r = 0; r < rows; ++r) out[(long long)c * rows + r] = in[(long long)r * cols + c];
}

// ---------------------------------------------------------------------------
// bf16 WMMA GEMM: C[M,N] = A[M,Kd] * B[Kd,N]  (A,B row-major bf16)
// Block tile 128x64, 256 threads = 8 waves in a 4x2 grid; each wave owns a
// 32x32 patch -> 4 accumulators, 4 v_wmma_f32_16x16x32_bf16 per K-step.
// A tile staged with GLOBAL_LOAD_ASYNC_TO_LDS_B128 (ASYNCcnt), B tile staged
// via VGPRs with an LDS transpose. Double-buffered LDS, one barrier per step.
// Output type is a template parameter -> branch-free store epilogue.
// ---------------------------------------------------------------------------
#define SA_STRIDE 48   // ushorts per row (128 rows) -> 96B rows, 16B aligned
#define SB_STRIDE 40   // ushorts per row (64 rows, transposed tile: [n][k])
#define SA_BYTES  (128 * SA_STRIDE * 2)
#define SB_BYTES  (64 * SB_STRIDE * 2)

template <bool OUT_BF16>
__global__ __launch_bounds__(256) void tfc_gemm_bf16_kernel(
    const unsigned short* __restrict__ A,
    const unsigned short* __restrict__ B,
    void* __restrict__ C,
    int M, int N, int Kd) {

  __shared__ unsigned short sA[2][128 * SA_STRIDE];
  __shared__ unsigned short sB[2][64 * SB_STRIDE];

  const int tid  = threadIdx.x;
  const int lane = tid & 31;
  const int wave = tid >> 5;
  const int wr   = wave >> 1;                // 0..3 -> 32-row band
  const int wc   = wave & 1;                 // 0..1 -> 32-col band
  const int rowBase = blockIdx.x * 128;
  const int colBase = blockIdx.y * 64;

  // A staging: each thread async-copies 32B (two B128 ops) of one row
  const int arow  = tid >> 1;                // 0..127
  const int ahalf = (tid & 1) * 16;          // element offset 0 / 16
  // B staging: each thread loads 8 bf16 of one B row, writes transposed
  const int bk = tid >> 3;                   // 0..31  (K within tile)
  const int bn = (tid & 7) * 8;              // 0..56  (N within tile)

  // WMMA fragment lane mapping (ISA 7.12.2)
  const int fm     = lane & 15;
  const int aKhalf = (lane < 16) ? 0 : 8;
  const int bKlo   = (lane < 16) ? 0 : 8;

  // LDS byte offsets (flat LDS address carries DS offset in low 32 bits)
  const unsigned sAoff0 = (unsigned)(size_t)&sA[0][0];
  const unsigned aLds   = sAoff0 + (unsigned)(arow * SA_STRIDE + ahalf) * 2;

  AccF acc[2][2];
#pragma unroll
  for (int rs = 0; rs < 2; ++rs)
#pragma unroll
    for (int cs = 0; cs < 2; ++cs)
#pragma unroll
      for (int v = 0; v < 8; ++v) acc[rs][cs].f[v] = 0.f;

  union BReg { uint4 u4; unsigned short s[8]; } breg;

  // ---- prologue: stage tile 0 into buffer 0 ----
  {
    const unsigned short* gp = A + (long long)(rowBase + arow) * Kd + ahalf;
    unsigned l0 = aLds;
    asm volatile("global_load_async_to_lds_b128 %0, %1, off"
                 :: "v"(l0), "v"(gp) : "memory");
    asm volatile("global_load_async_to_lds_b128 %0, %1, off"
                 :: "v"(l0 + 16u), "v"(gp + 8) : "memory");
    breg.u4 = *(const uint4*)(B + (long long)bk * N + colBase + bn);
#pragma unroll
    for (int j = 0; j < 8; ++j) sB[0][(bn + j) * SB_STRIDE + bk] = breg.s[j];
  }
  asm volatile("s_wait_asynccnt 0" ::: "memory");
  __syncthreads();

  int buf = 0;
  for (int k0 = 0; k0 < Kd; k0 += 32) {
    const int nxt = k0 + 32;
    const bool has_next = (nxt < Kd);

    // ---- prefetch next tile into the other buffer ----
    if (has_next) {
      const unsigned short* gp = A + (long long)(rowBase + arow) * Kd + nxt + ahalf;
      unsigned l0 = aLds + (unsigned)((buf ^ 1) ? SA_BYTES : 0);
      asm volatile("global_load_async_to_lds_b128 %0, %1, off"
                   :: "v"(l0), "v"(gp) : "memory");
      asm volatile("global_load_async_to_lds_b128 %0, %1, off"
                   :: "v"(l0 + 16u), "v"(gp + 8) : "memory");
      breg.u4 = *(const uint4*)(B + (long long)(nxt + bk) * N + colBase + bn);
    }

    // ---- fragments from current buffer ----
    FragBF afr[2], bfr[2];
#pragma unroll
    for (int rs = 0; rs < 2; ++rs) {
      const int rowL = wr * 32 + rs * 16 + fm;
#pragma unroll
      for (int p = 0; p < 8; ++p) {
        int kk = (p < 4) ? (aKhalf + 2 * p) : (16 + aKhalf + 2 * (p - 4));
        afr[rs].u[p] = *(const unsigned int*)&sA[buf][rowL * SA_STRIDE + kk];
      }
    }
#pragma unroll
    for (int cs = 0; cs < 2; ++cs) {
      const int colL = wc * 32 + cs * 16 + fm;
#pragma unroll
      for (int p = 0; p < 8; ++p)
        bfr[cs].u[p] = *(const unsigned int*)&sB[buf][colL * SB_STRIDE + ((p < 4) ? (bKlo + 2 * p) : (16 + bKlo + 2 * (p - 4)))];
    }
#pragma unroll
    for (int rs = 0; rs < 2; ++rs)
#pragma unroll
      for (int cs = 0; cs < 2; ++cs)
        acc[rs][cs].v = __builtin_amdgcn_wmma_f32_16x16x32_bf16(
            false, afr[rs].v, false, bfr[cs].v, (short)0, acc[rs][cs].v, false, false);

    // ---- finish staging next B tile (transposed scatter) ----
    if (has_next) {
#pragma unroll
      for (int j = 0; j < 8; ++j) sB[buf ^ 1][(bn + j) * SB_STRIDE + bk] = breg.s[j];
    }
    // our async writes must land before anyone reads past the barrier
    asm volatile("s_wait_asynccnt 0" ::: "memory");
    __syncthreads();
    buf ^= 1;
  }

  // ---- store C (f32 frag: lane<16 -> M=v, lane>=16 -> M=8+v; N = lane&15) ----
  // Branch-free epilogue per instantiation; one base offset per (rs,cs).
  const int mOff = (lane < 16) ? 0 : 8;
  const int rTop = rowBase + wr * 32 + mOff;
  const int cTop = colBase + wc * 32 + fm;
#pragma unroll
  for (int rs = 0; rs < 2; ++rs) {
#pragma unroll
    for (int cs = 0; cs < 2; ++cs) {
      long long base = (long long)(rTop + rs * 16) * N + (cTop + cs * 16);
      if (OUT_BF16) {
        unsigned short* cp = (unsigned short*)C + base;
#pragma unroll
        for (int v = 0; v < 8; ++v) cp[(long long)v * N] = f32_to_bf16(acc[rs][cs].f[v]);
      } else {
        float* cp = (float*)C + base;
#pragma unroll
        for (int v = 0; v < 8; ++v) cp[(long long)v * N] = acc[rs][cs].f[v];
      }
    }
  }
}

// ---------------------------------------------------------------------------
// Sequential scan: one block per batch b, 256 threads = one mode k each.
// Writes raw real states as bf16; applies R to the chunk-final state for the
// carry (LDS broadcast + coalesced column reads of R).
// ---------------------------------------------------------------------------
__global__ __launch_bounds__(256) void tfc_scan_kernel(
    const float* __restrict__ beta,           // (B,T,K) f32
    const float* __restrict__ alpha_raw,      // (K,)
    const float* __restrict__ omega,          // (K,)
    const float* __restrict__ Rf,             // (K,K) f32
    unsigned short* __restrict__ states) {    // (B,T,K) bf16 raw real states

  __shared__ float lr[KK];
  __shared__ float li[KK];

  const int k = threadIdx.x;
  const int b = blockIdx.x;

  const float a   = alpha_raw[k];
  const float mag = 0.1f + (1.0f / (1.0f + expf(-a))) * (0.99f - 0.1f);
  const float ph  = omega[k] * 0.1f;          // omega * DT
  const float cp  = cosf(ph);
  const float sp  = sinf(ph);

  float r = 0.f, i = 0.f;

  for (int c = 0; c < NCHUNK; ++c) {
    long long base = ((long long)b * TT + (long long)c * CHUNKN) * KK + k;
    for (int t = 0; t < CHUNKN; ++t) {
      float bt = beta[base + (long long)t * KK];
      float nr = fmaf(mag, r * cp - i * sp, bt);   // beta enters real part only
      float ni = mag * (r * sp + i * cp);
      r = nr; i = ni;
      states[base + (long long)t * KK] = f32_to_bf16(nr);
    }
    // carry = (last raw state) @ R  (== (chunk @ R) last row)
    lr[k] = r; li[k] = i;
    __syncthreads();
    float nr2 = 0.f, ni2 = 0.f;
    for (int j = 0; j < KK; ++j) {
      float rv = Rf[j * KK + k];                   // coalesced across k
      nr2 = fmaf(lr[j], rv, nr2);
      ni2 = fmaf(li[j], rv, ni2);
    }
    __syncthreads();
    r = nr2; i = ni2;
  }
}

// ---------------------------------------------------------------------------
// Host-side launch (graph-capture safe: only kernel launches on `stream`)
// ---------------------------------------------------------------------------
extern "C" void kernel_launch(void* const* d_in, const int* in_sizes, int n_in,
                              void* d_out, int out_size, void* d_ws, size_t ws_size,
                              hipStream_t stream) {
  (void)in_sizes; (void)n_in; (void)out_size; (void)ws_size;

  const float* x         = (const float*)d_in[0];   // (B,T,D)
  const float* alpha_raw = (const float*)d_in[1];   // (K,)
  const float* omega     = (const float*)d_in[2];   // (K,)
  const float* W_in      = (const float*)d_in[3];   // (K,D)
  const float* W_out     = (const float*)d_in[4];   // (D,K)
  const float* Mm        = (const float*)d_in[5];   // (K,K)
  float* out             = (float*)d_out;           // (B,T,D) f32

  // ---- workspace carve-up (256B aligned) ----
  char* ws = (char*)d_ws;
  size_t off = 0;
  auto carve = [&](size_t bytes) -> char* {
    char* p = ws + off;
    off += (bytes + 255) & ~(size_t)255;
    return p;
  };
  unsigned short* x_bf    = (unsigned short*)carve((size_t)MROWS * DD * 2);  // 64 MB
  float*          beta    = (float*)         carve((size_t)MROWS * KK * 4);  // 32 MB
  unsigned short* st_bf   = (unsigned short*)carve((size_t)MROWS * KK * 2);  // 16 MB
  unsigned short* stR_bf  = (unsigned short*)carve((size_t)MROWS * KK * 2);  // 16 MB
  unsigned short* WinT    = (unsigned short*)carve((size_t)DD * KK * 2);
  unsigned short* WoutT   = (unsigned short*)carve((size_t)KK * DD * 2);
  float*          Rf      = (float*)         carve((size_t)KK * KK * 4);
  unsigned short* Rb      = (unsigned short*)carve((size_t)KK * KK * 2);
  float*          normBuf = (float*)         carve(256);

  // 1) ||M||_F, then R (f32 + bf16)
  tfc_norm_kernel<<<1, 256, 0, stream>>>(Mm, normBuf);
  tfc_buildR_kernel<<<(KK * KK) / 256, 256, 0, stream>>>(Mm, normBuf, Rf, Rb);

  // 2) precision conversions
  tfc_cvt_kernel<<<2048, 256, 0, stream>>>(x, x_bf, (long long)MROWS * DD);
  tfc_cvtT_kernel<<<DD / 256, 256, 0, stream>>>(W_in, WinT, KK, DD);   // WinT[d*K+k]
  tfc_cvtT_kernel<<<KK / 256, 256, 0, stream>>>(W_out, WoutT, DD, KK); // WoutT[k*D+d]

  // 3) beta = x @ W_in^T : (32768 x 1024) * (1024 x 256) -> f32
  {
    dim3 grid(MROWS / 128, KK / 64);
    tfc_gemm_bf16_kernel<false><<<grid, 256, 0, stream>>>(x_bf, WinT, beta,
                                                          MROWS, KK, DD);
  }

  // 4) sequential oscillator scan with R-coupled chunk carries
  tfc_scan_kernel<<<BB, KK, 0, stream>>>(beta, alpha_raw, omega, Rf, st_bf);

  // 5) statesR = states @ R : (32768 x 256) * (256 x 256) -> bf16
  {
    dim3 grid(MROWS / 128, KK / 64);
    tfc_gemm_bf16_kernel<true><<<grid, 256, 0, stream>>>(st_bf, Rb, stR_bf,
                                                         MROWS, KK, KK);
  }

  // 6) out = statesR @ W_out^T : (32768 x 256) * (256 x 1024) -> f32
  {
    dim3 grid(MROWS / 128, DD / 64);
    tfc_gemm_bf16_kernel<false><<<grid, 256, 0, stream>>>(stR_bf, WoutT, out,
                                                          MROWS, DD, KK);
  }
}
